// MultiHeadSelfAttention_18373870092947
// MI455X (gfx1250) — matrix operand order repaired and checked by reading, not compiled, not run on hardware
//
#include <hip/hip_runtime.h>
#include <hip/hip_bf16.h>

typedef __attribute__((ext_vector_type(16))) _Float16 v16h;
typedef __attribute__((ext_vector_type(8)))  _Float16 v8h;
typedef __attribute__((ext_vector_type(4)))  _Float16 v4h;
typedef __attribute__((ext_vector_type(8)))  float    v8f;

#define HIDDEN 1024
#define NHEADS 16
#define HDIM   64
#define SEQ    2048
#define BATCH  2
#define NTOK   (BATCH * SEQ)   // 4096
#define LN_EPS 1e-6f
#define SM_SCALE 0.03125f      // 1/sqrt(1024)
#define LDS_STRIDE 40          // padded row stride (halfs): 80B = 20 dwords -> conflict-free

// ---------------------------------------------------------------------------
// WMMA helpers (CDNA5, wave32): D(16x16,f32) = A(16x32,f16) * B(32x16,f16) + C
// ---------------------------------------------------------------------------
__device__ __forceinline__ v8f wmma_16x16x32(v16h a, v16h b, v8f c) {
  return __builtin_amdgcn_wmma_f32_16x16x32_f16(
      /*neg_a=*/false, a, /*neg_b=*/false, b,
      /*c_mod=*/(short)0, c, /*reuse_a=*/false, /*reuse_b=*/false);
}

// A-fragment (16x32 f16). Lane holds row M = lane%16.
// halfs[0..7]  = row[k0 + base + 0..7]
// halfs[8..15] = row[k0 + base + 16..23],  base = (lane>=16) ? 8 : 0
__device__ __forceinline__ v16h load_fragA(const _Float16* row, int k0, int lane) {
  int base = ((lane >> 4) & 1) * 8;
  v8h lo = *(const v8h*)(row + k0 + base);
  v8h hi = *(const v8h*)(row + k0 + base + 16);
  v16h r;
#pragma unroll
  for (int i = 0; i < 8; ++i) { r[i] = lo[i]; r[i + 8] = hi[i]; }
  return r;
}

// B-fragment (32x16 f16). Lane holds column N = lane%16; same K order as A:
// halfs[0..7] = K[base+0..7], halfs[8..15] = K[base+16..23], base = (lane>=16) ? 8 : 0.
__device__ __forceinline__ v16h load_fragB(const _Float16* row, int k0, int lane) {
  int base = ((lane >> 4) & 1) * 8;
  v8h lo = *(const v8h*)(row + k0 + base);
  v8h hi = *(const v8h*)(row + k0 + base + 16);
  v16h r;
#pragma unroll
  for (int i = 0; i < 8; ++i) { r[i] = lo[i]; r[i + 8] = hi[i]; }
  return r;
}

// Async copy of one 16-byte chunk global -> LDS (tracked by ASYNCcnt).
__device__ __forceinline__ void async_copy_b128(const _Float16* gsrc, unsigned lds_off) {
  asm volatile("global_load_async_to_lds_b128 %0, %1, off"
               :: "v"(lds_off), "v"(gsrc) : "memory");
}

// ---------------------------------------------------------------------------
// Kernel 1: LayerNorm. One 256-thread block per token row (4 floats/thread).
// ---------------------------------------------------------------------------
__global__ __launch_bounds__(256)
void mhsa_ln_kernel(const float* __restrict__ x, const float* __restrict__ g,
                    const float* __restrict__ bb, float* __restrict__ xn32,
                    _Float16* __restrict__ xnh) {
  __shared__ float red[256];
  const int row = blockIdx.x;
  const int t = threadIdx.x;
  const float* xr = x + (size_t)row * HIDDEN;
  float4 v = ((const float4*)xr)[t];

  red[t] = v.x + v.y + v.z + v.w;
  __syncthreads();
  for (int off = 128; off > 0; off >>= 1) {
    if (t < off) red[t] += red[t + off];
    __syncthreads();
  }
  const float mu = red[0] * (1.0f / HIDDEN);
  __syncthreads();

  float d0 = v.x - mu, d1 = v.y - mu, d2 = v.z - mu, d3 = v.w - mu;
  red[t] = d0 * d0 + d1 * d1 + d2 * d2 + d3 * d3;
  __syncthreads();
  for (int off = 128; off > 0; off >>= 1) {
    if (t < off) red[t] += red[t + off];
    __syncthreads();
  }
  const float rstd = rsqrtf(red[0] * (1.0f / HIDDEN) + LN_EPS);

  float4 gv = ((const float4*)g)[t];
  float4 bv = ((const float4*)bb)[t];
  float4 o;
  o.x = d0 * rstd * gv.x + bv.x;
  o.y = d1 * rstd * gv.y + bv.y;
  o.z = d2 * rstd * gv.z + bv.z;
  o.w = d3 * rstd * gv.w + bv.w;
  ((float4*)(xn32 + (size_t)row * HIDDEN))[t] = o;
  v4h h;
  h[0] = (_Float16)o.x; h[1] = (_Float16)o.y;
  h[2] = (_Float16)o.z; h[3] = (_Float16)o.w;
  *(v4h*)(xnh + (size_t)row * HIDDEN + t * 4) = h;
}

// ---------------------------------------------------------------------------
// Kernel 2: cast + transpose weights: wT[mat][n][k] = (f16) w[mat][k][n]
// ---------------------------------------------------------------------------
__global__ __launch_bounds__(256)
void mhsa_wcast_kernel(const float* __restrict__ wq, const float* __restrict__ wk,
                       const float* __restrict__ wv, _Float16* __restrict__ wT) {
  int idx = blockIdx.x * 256 + threadIdx.x;
  if (idx >= 3 * HIDDEN * HIDDEN) return;
  int mat = idx >> 20;
  int k = (idx >> 10) & 1023;
  int n = idx & 1023;
  const float* w = (mat == 0) ? wq : ((mat == 1) ? wk : wv);
  wT[((size_t)mat << 20) + ((size_t)n << 10) + k] = (_Float16)w[(k << 10) + n];
}

// ---------------------------------------------------------------------------
// Kernel 3: QKV projection GEMM. M=4096 (tokens), N=1024, K=1024.
// Block = 8 waves (4 along M x 2 along N); wave tile = 32x64 (2x4 WMMA accs).
// Operand slabs (A 128x32, B 128x32 per k-step) are shared by the block, so
// they are staged into double-buffered LDS with GLOBAL_LOAD_ASYNC_TO_LDS_B128
// (ASYNCcnt), overlapping the next stage's copies with the current WMMAs.
// z = 0/1: Q/K -> head-split [b][h][s][d] f16
// z = 2  : V  -> transposed  [b][h][d][s] f16 (packed 16B stores)
// ---------------------------------------------------------------------------
__global__ __launch_bounds__(256)
void mhsa_qkv_gemm_kernel(const _Float16* __restrict__ A, const _Float16* __restrict__ wT,
                          _Float16* __restrict__ Qb, _Float16* __restrict__ Kb,
                          _Float16* __restrict__ Vt) {
  __shared__ _Float16 Atile[2][128 * LDS_STRIDE];
  __shared__ _Float16 Btile[2][128 * LDS_STRIDE];

  const int z = blockIdx.z;
  const _Float16* W = wT + ((size_t)z << 20);
  const int lane = threadIdx.x & 31;
  const int wave = threadIdx.x >> 5;
  const int MoBlk = blockIdx.x * 128;
  const int NoBlk = blockIdx.y * 128;
  const int wm = wave & 3, wn = wave >> 2;
  const int lr = lane & 15;
  const int hiRow = ((lane >> 4) & 1) * 8;

  // per-thread staging assignment: 2 chunks of A + 2 chunks of B per k-step
  int srow[2], sck[2];
#pragma unroll
  for (int i = 0; i < 2; ++i) {
    int c = threadIdx.x + i * 256;        // 0..511
    srow[i] = c >> 2;                     // 0..127
    sck[i] = (c & 3) * 8;                 // half offset within 32-wide slab
  }

  v8f acc[2][4];
#pragma unroll
  for (int mi = 0; mi < 2; ++mi)
#pragma unroll
    for (int ni = 0; ni < 4; ++ni)
#pragma unroll
      for (int e = 0; e < 8; ++e) acc[mi][ni][e] = 0.0f;

  // prologue: stage k0 = 0 into buffer 0  (4 async issues per thread)
#pragma unroll
  for (int i = 0; i < 2; ++i) {
    async_copy_b128(A + (size_t)(MoBlk + srow[i]) * HIDDEN + sck[i],
                    (unsigned)(uintptr_t)&Atile[0][srow[i] * LDS_STRIDE + sck[i]]);
    async_copy_b128(W + (size_t)(NoBlk + srow[i]) * HIDDEN + sck[i],
                    (unsigned)(uintptr_t)&Btile[0][srow[i] * LDS_STRIDE + sck[i]]);
  }

  for (int k0 = 0; k0 < HIDDEN; k0 += 32) {
    const int buf = (k0 >> 5) & 1;
    if (k0 + 32 < HIDDEN) {
      // stage next k-slab into the other buffer while we compute this one
      const int nk = k0 + 32, nb = buf ^ 1;
#pragma unroll
      for (int i = 0; i < 2; ++i) {
        async_copy_b128(A + (size_t)(MoBlk + srow[i]) * HIDDEN + nk + sck[i],
                        (unsigned)(uintptr_t)&Atile[nb][srow[i] * LDS_STRIDE + sck[i]]);
        async_copy_b128(W + (size_t)(NoBlk + srow[i]) * HIDDEN + nk + sck[i],
                        (unsigned)(uintptr_t)&Btile[nb][srow[i] * LDS_STRIDE + sck[i]]);
      }
      asm volatile("s_wait_asynccnt 0x4" ::: "memory");  // current slab landed
    } else {
      asm volatile("s_wait_asynccnt 0x0" ::: "memory");  // drain final slab
    }
    __syncthreads();  // all waves' staged data visible

    v16h aF[2], bF[4];
#pragma unroll
    for (int mi = 0; mi < 2; ++mi)
      aF[mi] = load_fragA(&Atile[buf][(wm * 32 + mi * 16 + lr) * LDS_STRIDE], 0, lane);
#pragma unroll
    for (int ni = 0; ni < 4; ++ni)
      bF[ni] = load_fragB(&Btile[buf][(wn * 64 + ni * 16 + lr) * LDS_STRIDE], 0, lane);
#pragma unroll
    for (int mi = 0; mi < 2; ++mi)
#pragma unroll
      for (int ni = 0; ni < 4; ++ni)
        acc[mi][ni] = wmma_16x16x32(aF[mi], bF[ni], acc[mi][ni]);

    __syncthreads();  // protect this buffer from the next stage's async writes
  }

  const int Mo = MoBlk + wm * 32;
  const int No = NoBlk + wn * 64;
  if (z < 2) {
    _Float16* dst = (z == 0) ? Qb : Kb;
#pragma unroll
    for (int mi = 0; mi < 2; ++mi)
#pragma unroll
      for (int ni = 0; ni < 4; ++ni) {
        const int ng = No + ni * 16 + lr;
        const int h = ng >> 6, d = ng & 63;
#pragma unroll
        for (int r = 0; r < 8; ++r) {
          const int mg = Mo + mi * 16 + r + hiRow;
          const int b = mg >> 11, s = mg & 2047;
          dst[(((size_t)b * NHEADS + h) * SEQ + s) * HDIM + d] = (_Float16)acc[mi][ni][r];
        }
      }
  } else {
#pragma unroll
    for (int mi = 0; mi < 2; ++mi)
#pragma unroll
      for (int ni = 0; ni < 4; ++ni) {
        const int ng = No + ni * 16 + lr;
        const int h = ng >> 6, d = ng & 63;
        const int mg = Mo + mi * 16 + hiRow;
        const int b = mg >> 11, s = mg & 2047;
        v8h pk;
#pragma unroll
        for (int r = 0; r < 8; ++r) pk[r] = (_Float16)acc[mi][ni][r];
        *(v8h*)(Vt + (((size_t)b * NHEADS + h) * HDIM + d) * SEQ + s) = pk;
      }
  }
}

// ---------------------------------------------------------------------------
// Kernel 4: flash attention. One wave handles a 16-row query tile of one head.
// 4 waves/block, each with a private 16x32 f16 LDS tile for the P transpose.
// Per 32-key block: 4 WMMAs (Q*K^T) + online softmax + 4 WMMAs (P*V).
// ---------------------------------------------------------------------------
__global__ __launch_bounds__(128)
void mhsa_attn_kernel(const _Float16* __restrict__ Qb, const _Float16* __restrict__ Kb,
                      const _Float16* __restrict__ Vt, const float* __restrict__ xn32,
                      float* __restrict__ out) {
  __shared__ _Float16 Pbuf[4][16 * 32];
  const int lane = threadIdx.x & 31;
  const int wave = threadIdx.x >> 5;
  const int wg = blockIdx.x * 4 + wave;      // 0 .. 4095
  const int b = wg >> 11;
  const int rem = wg & 2047;
  const int h = rem >> 7;
  const int m0 = (rem & 127) << 4;
  const int lr = lane & 15;
  const int hiRow = ((lane >> 4) & 1) * 8;

  const _Float16* Q = Qb + (((size_t)b * NHEADS + h) * SEQ + m0) * HDIM;
  const _Float16* K = Kb + (((size_t)b * NHEADS + h) * SEQ) * HDIM;
  const _Float16* V = Vt + (((size_t)b * NHEADS + h) * HDIM) * SEQ;
  _Float16* P = Pbuf[wave];

  v16h qF[2];
  qF[0] = load_fragA(Q + (size_t)lr * HDIM, 0, lane);
  qF[1] = load_fragA(Q + (size_t)lr * HDIM, 32, lane);

  float rm[8], rs[8];
  v8f acc[4];
#pragma unroll
  for (int r = 0; r < 8; ++r) { rm[r] = -3.0e38f; rs[r] = 0.0f; }
#pragma unroll
  for (int c = 0; c < 4; ++c)
#pragma unroll
    for (int e = 0; e < 8; ++e) acc[c][e] = 0.0f;

  for (int n0 = 0; n0 < SEQ; n0 += 32) {
    if (n0 + 32 < SEQ) {  // hint next K block toward the WGP (global_prefetch_b8)
      __builtin_prefetch(K + (size_t)(n0 + 32 + lr) * HDIM, 0, 0);
      __builtin_prefetch(K + (size_t)(n0 + 48 + lr) * HDIM, 0, 0);
    }
    // ---- scores for 32 keys: two 16x16 tiles, each K-dim = 64 (2 steps) ----
    v8f st[2];
#pragma unroll
    for (int t = 0; t < 2; ++t) {
      const _Float16* Krow = K + (size_t)(n0 + t * 16 + lr) * HDIM;
      v16h kb0 = load_fragB(Krow, 0, lane);
      v16h kb1 = load_fragB(Krow, 32, lane);
      v8f zz;
#pragma unroll
      for (int e = 0; e < 8; ++e) zz[e] = 0.0f;
      zz = wmma_16x16x32(qF[0], kb0, zz);
      zz = wmma_16x16x32(qF[1], kb1, zz);
      st[t] = zz;
    }
#pragma unroll
    for (int t = 0; t < 2; ++t)
#pragma unroll
      for (int r = 0; r < 8; ++r) st[t][r] *= SM_SCALE;

    // ---- online softmax (row reductions across 16-lane halves) ----
    float alpha[8];
#pragma unroll
    for (int r = 0; r < 8; ++r) {
      float v = fmaxf(st[0][r], st[1][r]);
      v = fmaxf(v, __shfl_xor(v, 1, 32));
      v = fmaxf(v, __shfl_xor(v, 2, 32));
      v = fmaxf(v, __shfl_xor(v, 4, 32));
      v = fmaxf(v, __shfl_xor(v, 8, 32));
      const float nm = fmaxf(rm[r], v);
      alpha[r] = __expf(rm[r] - nm);
      rm[r] = nm;
    }
#pragma unroll
    for (int r = 0; r < 8; ++r) {
      const float p0 = __expf(st[0][r] - rm[r]);
      const float p1 = __expf(st[1][r] - rm[r]);
      st[0][r] = p0; st[1][r] = p1;
      float v = p0 + p1;
      v += __shfl_xor(v, 1, 32);
      v += __shfl_xor(v, 2, 32);
      v += __shfl_xor(v, 4, 32);
      v += __shfl_xor(v, 8, 32);
      rs[r] = rs[r] * alpha[r] + v;
    }
#pragma unroll
    for (int c = 0; c < 4; ++c)
#pragma unroll
      for (int r = 0; r < 8; ++r) acc[c][r] *= alpha[r];

    // ---- transpose P (D-layout -> A-layout) through LDS ----
    __syncthreads();   // WAR vs previous iteration's reads
#pragma unroll
    for (int t = 0; t < 2; ++t)
#pragma unroll
      for (int r = 0; r < 8; ++r)
        P[(r + hiRow) * 32 + t * 16 + lr] = (_Float16)st[t][r];
    __syncthreads();   // RAW: writes visible before reads
    v16h pA = load_fragA(P + lr * 32, 0, lane);

    // ---- out += P @ V (V transposed: contiguous along keys) ----
#pragma unroll
    for (int c = 0; c < 4; ++c) {
      const _Float16* Vrow = V + (size_t)(c * 16 + lr) * SEQ + n0;
      v16h vB = load_fragB(Vrow, 0, lane);
      acc[c] = wmma_16x16x32(pA, vB, acc[c]);
    }
  }

  // ---- epilogue: normalize + residual, write f32 output ----
#pragma unroll
  for (int c = 0; c < 4; ++c) {
    const int d = c * 16 + lr;
#pragma unroll
    for (int r = 0; r < 8; ++r) {
      const int row = r + hiRow;
      const size_t o = ((size_t)(b * SEQ + m0 + row)) * HIDDEN + h * HDIM + d;
      out[o] = acc[c][r] / rs[r] + xn32[o];
    }
  }
}

// ---------------------------------------------------------------------------
// Host launcher
// ---------------------------------------------------------------------------
extern "C" void kernel_launch(void* const* d_in, const int* in_sizes, int n_in,
                              void* d_out, int out_size, void* d_ws, size_t ws_size,
                              hipStream_t stream) {
  const float* x        = (const float*)d_in[0];
  const float* ln_scale = (const float*)d_in[1];
  const float* ln_bias  = (const float*)d_in[2];
  const float* wq       = (const float*)d_in[3];
  const float* wk       = (const float*)d_in[4];
  const float* wv       = (const float*)d_in[5];
  float* out = (float*)d_out;

  char* ws = (char*)d_ws;
  float*    xn32 = (float*)(ws);                               // 16 MB
  _Float16* xnh  = (_Float16*)(ws + 16777216);                 //  8 MB
  _Float16* wT   = (_Float16*)(ws + 16777216 + 8388608);       //  6 MB
  _Float16* Qb   = (_Float16*)(ws + 31457280);                 //  8 MB
  _Float16* Kb   = (_Float16*)(ws + 39845888);                 //  8 MB
  _Float16* Vt   = (_Float16*)(ws + 48234496);                 //  8 MB

  mhsa_ln_kernel<<<NTOK, 256, 0, stream>>>(x, ln_scale, ln_bias, xn32, xnh);
  mhsa_wcast_kernel<<<(3 * HIDDEN * HIDDEN) / 256, 256, 0, stream>>>(wq, wk, wv, wT);
  mhsa_qkv_gemm_kernel<<<dim3(NTOK / 128, HIDDEN / 128, 3), 256, 0, stream>>>(
      xnh, wT, Qb, Kb, Vt);
  mhsa_attn_kernel<<<dim3((BATCH * NHEADS * (SEQ / 16)) / 4), 128, 0, stream>>>(
      Qb, Kb, Vt, xn32, out);
}
